// TransformerSeqLayer_59047210385719
// MI455X (gfx1250) — matrix operand order repaired and checked by reading, not compiled, not run on hardware
//
#include <hip/hip_runtime.h>
#include <hip/hip_bf16.h>

// ---------------- problem constants (match reference) ----------------
constexpr int B_   = 4;
constexpr int M_   = 1024;
constexpr int H_   = 1024;
constexpr int NH_  = 8;
constexpr int D_   = 128;
constexpr int L_   = 1024;
constexpr int LM_  = L_ + M_;     // 2048
constexpr int E_   = 8;
constexpr int F_   = 4096;
constexpr int NTOK_ = B_ * M_;    // 4096
constexpr float MU_ = 0.7f;
constexpr float GAMMA_ = 1.0f;

typedef _Float16 f16;
typedef __attribute__((ext_vector_type(16))) _Float16 v16h;
typedef __attribute__((ext_vector_type(8)))  float    v8f;

static __device__ __forceinline__ v8f wmma_f16(v16h a, v16h b, v8f c) {
  // D = A(16x32 f16) * B(32x16 f16) + C(16x16 f32)
  return __builtin_amdgcn_wmma_f32_16x16x32_f16(false, a, false, b, (short)0, c,
                                                false, false);
}
static __device__ __forceinline__ v8f vzero8() {
  v8f z;
  #pragma unroll
  for (int i = 0; i < 8; ++i) z[i] = 0.f;
  return z;
}
// Build a v16h fragment from two 16B-aligned runs of 8 f16 each.
static __device__ __forceinline__ v16h load16(const f16* p0, const f16* p1) {
  union { uint4 u[2]; v16h h; } t;
  t.u[0] = *(const uint4*)p0;
  t.u[1] = *(const uint4*)p1;
  return t.h;
}
// Async global -> LDS copy of 16B per lane (CDNA5 path, ASYNCcnt-tracked).
// lds_off: LDS byte address (low 32 bits of the generic shared pointer).
static __device__ __forceinline__ void async_g2l_b128(unsigned lds_off,
                                                      const void* gaddr) {
  asm volatile("global_load_async_to_lds_b128 %0, %1, off"
               :
               : "v"(lds_off), "v"(gaddr)
               : "memory");
}
static __device__ __forceinline__ void wait_async0() {
  asm volatile("s_wait_asynccnt 0" ::: "memory");
}

// ---------------- generic fp32 -> f16 conversion ----------------
__global__ void cvt_kernel(const float* __restrict__ src, f16* __restrict__ dst,
                           size_t n) {
  size_t i = (size_t)blockIdx.x * blockDim.x + threadIdx.x;
  size_t stride = (size_t)gridDim.x * blockDim.x;
  for (; i < n; i += stride) dst[i] = (f16)src[i];
}

__global__ void zero_kernel(float* __restrict__ dst, size_t n) {
  size_t i = (size_t)blockIdx.x * blockDim.x + threadIdx.x;
  size_t stride = (size_t)gridDim.x * blockDim.x;
  for (; i < n; i += stride) dst[i] = 0.f;
}

// ---------------- WMMA GEMM ----------------
// C[Mr x Nc] = A[Mr x Kc] (f16, row-major) * B[Kc x Nc] (TB, row-major)
// Block: 256 threads = 8 waves. Block tile 128x128, wave tile 64x32,
// 4x2 fragments of 16x16, K-slab of 32 (one WMMA K).
// A tile is staged with async global->LDS b128 copies (no VGPR round trip);
// B tile goes through VGPRs (needs transpose + optional f32->f16 convert).
// Epilogues:
//   0: Ch = acc                         (f16 store)
//   1: Cf = acc + res[row*ldres+col]    (f32 store, residual add)
//   2: Ch = relu(acc + bias[col])       (f16 store)
//   3: Cf += rowscale[row*rs_stride] * (acc + bias[col])   (f32 accumulate)
//   4: Cf = acc                         (f32 store)
//   5: transposed f16 store into V^T layout [B*NH][D][LM]
template <int EPI, typename TB>
__global__ __launch_bounds__(256) void gemm_kernel(
    const f16* __restrict__ A, int lda, const TB* __restrict__ Bm, int ldb,
    float* __restrict__ Cf, f16* __restrict__ Ch, int ldc, int Mr, int Nc,
    int Kc, const float* __restrict__ bias, const float* __restrict__ res,
    int ldres, const float* __restrict__ rowscale, int rs_stride) {
  __shared__ __align__(16) f16 sA[128][32];
  __shared__ __align__(16) f16 sBT[128][40];  // [n][k], pad to 80B rows (16B mult)
  const int tid  = threadIdx.x;
  const int lane = tid & 31;
  const int wid  = tid >> 5;
  const int wr   = wid >> 2;       // 0..1  (row group of 64)
  const int wc   = wid & 3;        // 0..3  (col group of 32)
  const int nl   = lane & 15;
  const int hi   = lane >> 4;
  const int row0 = blockIdx.y * 128;
  const int col0 = blockIdx.x * 128;

  v8f acc[4][2];
  #pragma unroll
  for (int i = 0; i < 4; ++i)
    #pragma unroll
    for (int j = 0; j < 2; ++j) acc[i][j] = vzero8();

  for (int k0 = 0; k0 < Kc; k0 += 32) {
    // --- stage A tile 128x32 (f16): async global->LDS, 2x 16B per thread ---
    #pragma unroll
    for (int i = 0; i < 2; ++i) {
      int c  = tid + i * 256;      // chunk of 8 f16, 0..511
      int ar = c >> 2;             // 0..127
      int ac = (c & 3) * 8;
      async_g2l_b128((unsigned)(unsigned long long)&sA[ar][ac],
                     A + (size_t)(row0 + ar) * lda + k0 + ac);
    }
    // --- stage B tile 32x128, K-transposed into sBT[n][k] ---
    if constexpr (sizeof(TB) == 2) {
      #pragma unroll
      for (int i = 0; i < 2; ++i) {
        int c   = tid + i * 256;   // chunk of 8 f16, 0..511
        int kk  = c >> 4;          // 0..31
        int nn8 = (c & 15) * 8;
        union { uint4 u; f16 e[8]; } t;
        t.u = *(const uint4*)((const f16*)Bm + (size_t)(k0 + kk) * ldb + col0 +
                              nn8);
        #pragma unroll
        for (int j = 0; j < 8; ++j) sBT[nn8 + j][kk] = t.e[j];
      }
    } else {
      #pragma unroll
      for (int i = 0; i < 4; ++i) {
        int c   = tid + i * 256;   // chunk of 4 floats, 0..1023
        int kk  = c >> 5;          // 0..31
        int nn4 = (c & 31) * 4;
        float4 t = *(const float4*)((const float*)Bm +
                                    (size_t)(k0 + kk) * ldb + col0 + nn4);
        sBT[nn4 + 0][kk] = (f16)t.x;
        sBT[nn4 + 1][kk] = (f16)t.y;
        sBT[nn4 + 2][kk] = (f16)t.z;
        sBT[nn4 + 3][kk] = (f16)t.w;
      }
    }
    // --- prefetch next K-slab while we compute ---
    if (k0 + 32 < Kc) {
      __builtin_prefetch(A + (size_t)(row0 + (tid >> 2)) * lda + k0 + 32 +
                             (tid & 3) * 8, 0, 1);
      __builtin_prefetch((const char*)Bm +
                             ((size_t)(k0 + 32 + (tid >> 4)) * ldb + col0 +
                              (tid & 15) * 8) * sizeof(TB), 0, 1);
    }
    wait_async0();        // A tile resident (this wave's async copies done)
    __syncthreads();      // all waves' tiles visible

    // --- fragments (2x ds_load_b128 each) ---
    v16h af[4], bf[2];
    #pragma unroll
    for (int fr = 0; fr < 4; ++fr) {
      const f16* ap = &sA[wr * 64 + fr * 16 + nl][hi * 8];
      af[fr] = load16(ap, ap + 16);
    }
    #pragma unroll
    for (int fc = 0; fc < 2; ++fc) {
      const f16* bp = &sBT[wc * 32 + fc * 16 + nl][hi * 8];
      bf[fc] = load16(bp, bp + 16);
    }
    #pragma unroll
    for (int fr = 0; fr < 4; ++fr)
      #pragma unroll
      for (int fc = 0; fc < 2; ++fc)
        acc[fr][fc] = wmma_f16(af[fr], bf[fc], acc[fr][fc]);
    __syncthreads();
  }

  // --- epilogue ---
  #pragma unroll
  for (int fr = 0; fr < 4; ++fr) {
    #pragma unroll
    for (int fc = 0; fc < 2; ++fc) {
      #pragma unroll
      for (int r = 0; r < 8; ++r) {
        int row = row0 + wr * 64 + fr * 16 + hi * 8 + r;
        int col = col0 + wc * 32 + fc * 16 + nl;
        float v = acc[fr][fc][r];
        size_t ci = (size_t)row * ldc + col;
        if constexpr (EPI == 0) {
          Ch[ci] = (f16)v;
        } else if constexpr (EPI == 1) {
          Cf[ci] = v + res[(size_t)row * ldres + col];
        } else if constexpr (EPI == 2) {
          float t = v + bias[col];
          Ch[ci] = (f16)(t > 0.f ? t : 0.f);
        } else if constexpr (EPI == 3) {
          Cf[ci] += rowscale[(size_t)row * rs_stride] * (v + bias[col]);
        } else if constexpr (EPI == 4) {
          Cf[ci] = v;
        } else if constexpr (EPI == 5) {
          // row = b*LM + n (key pos), col = head*D + d  ->  V^T[bh][d][n]
          int bb = row / LM_, n = row % LM_;
          int hh = col / D_,  d = col % D_;
          Ch[(((size_t)(bb * NH_ + hh)) * D_ + d) * LM_ + n] = (f16)v;
        }
      }
    }
  }
}

// ---------------- sliding-window flash attention ----------------
// Query m attends keys [m, m+L-1] of h_all; score = (q.k + q.P[:,n-m])/sqrt(H)
// One wave per 16-query tile; online softmax over 32-key blocks.
__global__ __launch_bounds__(256) void attn_kernel(
    const f16* __restrict__ q,    // [B*M, H]
    const f16* __restrict__ k,    // [B*LM, H]
    const f16* __restrict__ vT,   // [B*NH, D, LM]
    const float* __restrict__ posq,  // [B*NH, M, L] (unscaled q.P)
    f16* __restrict__ outp)       // [B*M, H]
{
  __shared__ __align__(16) f16 sP[8][16][32];
  const int wid  = threadIdx.x >> 5;
  const int lane = threadIdx.x & 31;
  const int nl   = lane & 15;
  const int hi   = lane >> 4;
  const int QT   = M_ / 16;
  int wg   = blockIdx.x * 8 + wid;
  int qt   = wg % QT;
  int head = (wg / QT) % NH_;
  int b    = wg / (QT * NH_);
  int m0   = qt * 16;

  // Q fragments: 4 K-slabs of 32 over D=128 (2x global_load_b128 each)
  v16h qf[4];
  #pragma unroll
  for (int dc = 0; dc < 4; ++dc) {
    const f16* base = q + ((size_t)(b * M_ + m0 + nl)) * H_ + head * D_ +
                      dc * 32 + hi * 8;
    qf[dc] = load16(base, base + 16);
  }
  v8f o[8];
  #pragma unroll
  for (int i = 0; i < 8; ++i) o[i] = vzero8();
  float rmax[8], rsum[8];
  #pragma unroll
  for (int r = 0; r < 8; ++r) { rmax[r] = -1e30f; rsum[r] = 0.f; }

  const float isq = 0.03125f;  // 1/sqrt(H)
  size_t bh = (size_t)(b * NH_ + head);
  const float* posb = posq + bh * (size_t)M_ * L_;
  const f16* vb = vT + bh * (size_t)D_ * LM_;
  const f16* kbase = k + (size_t)(b * LM_) * H_ + head * D_;

  #pragma unroll 1
  for (int n0 = m0; n0 < m0 + L_ + 16; n0 += 32) {
    float pvs[2][8];
    // scores for two 16-key subtiles
    #pragma unroll
    for (int sub = 0; sub < 2; ++sub) {
      int key  = n0 + sub * 16 + nl;
      int keyc = key < LM_ ? key : LM_ - 1;
      v8f s = vzero8();
      #pragma unroll
      for (int dc = 0; dc < 4; ++dc) {
        const f16* kp = kbase + (size_t)keyc * H_ + dc * 32 + hi * 8;
        s = wmma_f16(qf[dc], load16(kp, kp + 16), s);
      }
      #pragma unroll
      for (int r = 0; r < 8; ++r) {
        int m = m0 + hi * 8 + r;
        int n = n0 + sub * 16 + nl;
        bool valid = (n >= m) && (n < m + L_) && (n < LM_);
        float val = -1e30f;
        if (valid)
          val = (s[r] + posb[(size_t)m * L_ + (n - m)]) * isq;
        pvs[sub][r] = val;
      }
    }
    // combined online-softmax update for the 32-key block
    #pragma unroll
    for (int r = 0; r < 8; ++r) {
      float tm = fmaxf(pvs[0][r], pvs[1][r]);
      #pragma unroll
      for (int off = 1; off < 16; off <<= 1)
        tm = fmaxf(tm, __shfl_xor(tm, off, 16));
      float nm = fmaxf(rmax[r], tm);
      float scale, p0, p1;
      if (nm <= -1e29f) {  // fully masked so far
        scale = 1.f; p0 = 0.f; p1 = 0.f;
      } else {
        scale = __expf(rmax[r] - nm);
        p0 = (pvs[0][r] <= -1e29f) ? 0.f : __expf(pvs[0][r] - nm);
        p1 = (pvs[1][r] <= -1e29f) ? 0.f : __expf(pvs[1][r] - nm);
      }
      rmax[r] = nm;
      float ps = p0 + p1;
      #pragma unroll
      for (int off = 1; off < 16; off <<= 1) ps += __shfl_xor(ps, off, 16);
      rsum[r] = rsum[r] * scale + ps;
      #pragma unroll
      for (int d8 = 0; d8 < 8; ++d8) o[d8][r] = o[d8][r] * scale;
      sP[wid][hi * 8 + r][nl]      = (f16)p0;
      sP[wid][hi * 8 + r][16 + nl] = (f16)p1;
    }
    asm volatile("s_wait_dscnt 0" ::: "memory");
    // P fragment (16 queries x 32 keys)
    const f16* pp = &sP[wid][nl][hi * 16];
    v16h pa = load16(pp, pp + 8);
    // O += P @ V   (V^T gives contiguous key runs per d)
    #pragma unroll
    for (int d8 = 0; d8 < 8; ++d8) {
      int kk0 = n0 + hi * 16;
      if (kk0 > LM_ - 16) kk0 = LM_ - 16;  // clamp; masked keys have p=0
      const f16* vp = vb + (size_t)(d8 * 16 + nl) * LM_ + kk0;
      o[d8] = wmma_f16(pa, load16(vp, vp + 8), o[d8]);
    }
  }
  // normalize + store
  #pragma unroll
  for (int r = 0; r < 8; ++r) {
    float inv = rsum[r] > 0.f ? 1.f / rsum[r] : 0.f;
    int m = m0 + hi * 8 + r;
    #pragma unroll
    for (int d8 = 0; d8 < 8; ++d8) {
      outp[((size_t)(b * M_ + m)) * H_ + head * D_ + d8 * 16 + nl] =
          (f16)(o[d8][r] * inv);
    }
  }
}

// ---------------- layernorm 1 ----------------
__global__ __launch_bounds__(256) void ln1_kernel(
    const float* __restrict__ pre, const float* __restrict__ g,
    const float* __restrict__ bta, float* __restrict__ out32,
    f16* __restrict__ out16) {
  int row = blockIdx.x, tid = threadIdx.x;
  float t[H_ / 256];
  float sum = 0.f, sq = 0.f;
  #pragma unroll
  for (int i = 0; i < H_ / 256; ++i) {
    int h = tid + i * 256;
    float v = pre[(size_t)row * H_ + h];
    t[i] = v; sum += v; sq += v * v;
  }
  __shared__ float r1[256], r2[256];
  r1[tid] = sum; r2[tid] = sq;
  __syncthreads();
  for (int s = 128; s > 0; s >>= 1) {
    if (tid < s) { r1[tid] += r1[tid + s]; r2[tid] += r2[tid + s]; }
    __syncthreads();
  }
  float mean = r1[0] / H_;
  float var  = r2[0] / H_ - mean * mean;
  float rs   = rsqrtf(var + 1e-5f);
  #pragma unroll
  for (int i = 0; i < H_ / 256; ++i) {
    int h = tid + i * 256;
    float y = (t[i] - mean) * rs * g[h] + bta[h];
    out32[(size_t)row * H_ + h] = y;
    out16[(size_t)row * H_ + h] = (f16)y;
  }
}

// ---------------- top-2 gate ----------------
__global__ __launch_bounds__(256) void gate_kernel(
    const float* __restrict__ x, const float* __restrict__ gw,
    const float* __restrict__ gb, float* __restrict__ gate) {
  int row = blockIdx.x, tid = threadIdx.x;
  float part[E_];
  #pragma unroll
  for (int e = 0; e < E_; ++e) part[e] = 0.f;
  for (int h = tid; h < H_; h += 256) {
    float xv = x[(size_t)row * H_ + h];
    #pragma unroll
    for (int e = 0; e < E_; ++e) part[e] += xv * gw[h * E_ + e];
  }
  __shared__ float red[256];
  __shared__ float logits[E_];
  for (int e = 0; e < E_; ++e) {
    red[tid] = part[e];
    __syncthreads();
    for (int s = 128; s > 0; s >>= 1) {
      if (tid < s) red[tid] += red[tid + s];
      __syncthreads();
    }
    if (tid == 0) logits[e] = red[0] + gb[e];
    __syncthreads();
  }
  if (tid == 0) {
    int i1 = 0; float v1 = logits[0];
    for (int e = 1; e < E_; ++e)
      if (logits[e] > v1) { v1 = logits[e]; i1 = e; }
    int i2 = -1; float v2 = -1e30f;
    for (int e = 0; e < E_; ++e)
      if (e != i1 && logits[e] > v2) { v2 = logits[e]; i2 = e; }
    float be = __expf(v2 - v1);
    float s  = 1.f + be;
    for (int e = 0; e < E_; ++e) gate[(size_t)row * E_ + e] = 0.f;
    gate[(size_t)row * E_ + i1] = 1.f / s;
    gate[(size_t)row * E_ + i2] = be / s;
  }
}

// ---------------- momentum + layernorm 2 (final) ----------------
__global__ __launch_bounds__(256) void final_kernel(
    const float* __restrict__ h1, const float* __restrict__ momentum,
    const float* __restrict__ moe, const float* __restrict__ g,
    const float* __restrict__ bta, float* __restrict__ out_h2,
    float* __restrict__ out_mom) {
  int row = blockIdx.x, tid = threadIdx.x;
  float t[H_ / 256];
  float sum = 0.f, sq = 0.f;
  #pragma unroll
  for (int i = 0; i < H_ / 256; ++i) {
    int h = tid + i * 256;
    size_t idx = (size_t)row * H_ + h;
    float mom = MU_ * momentum[idx] + GAMMA_ * moe[idx];
    out_mom[idx] = mom;
    float tv = h1[idx] - mom;
    t[i] = tv; sum += tv; sq += tv * tv;
  }
  __shared__ float r1[256], r2[256];
  r1[tid] = sum; r2[tid] = sq;
  __syncthreads();
  for (int s = 128; s > 0; s >>= 1) {
    if (tid < s) { r1[tid] += r1[tid + s]; r2[tid] += r2[tid + s]; }
    __syncthreads();
  }
  float mean = r1[0] / H_;
  float var  = r2[0] / H_ - mean * mean;
  float rs   = rsqrtf(var + 1e-5f);
  #pragma unroll
  for (int i = 0; i < H_ / 256; ++i) {
    int h = tid + i * 256;
    out_h2[(size_t)row * H_ + h] = (t[i] - mean) * rs * g[h] + bta[h];
  }
}

// ---------------- host orchestration ----------------
extern "C" void kernel_launch(void* const* d_in, const int* in_sizes, int n_in,
                              void* d_out, int out_size, void* d_ws,
                              size_t ws_size, hipStream_t stream) {
  const float* h        = (const float*)d_in[0];
  const float* h_cache  = (const float*)d_in[1];
  const float* pos_enc  = (const float*)d_in[2];
  const float* momentum = (const float*)d_in[3];
  const float* Wq = (const float*)d_in[4];
  const float* Wk = (const float*)d_in[5];
  const float* Wv = (const float*)d_in[6];
  const float* Wo = (const float*)d_in[7];
  const float* gate_w = (const float*)d_in[8];
  const float* gate_b = (const float*)d_in[9];
  const float* w1 = (const float*)d_in[10];
  const float* b1 = (const float*)d_in[11];
  const float* w2 = (const float*)d_in[12];
  const float* b2 = (const float*)d_in[13];
  const float* ln1_g = (const float*)d_in[14];
  const float* ln1_b = (const float*)d_in[15];
  const float* ln2_g = (const float*)d_in[16];
  const float* ln2_b = (const float*)d_in[17];

  char* wsp = (char*)d_ws;
  size_t off = 0;
  auto alloc = [&](size_t bytes) -> char* {
    char* p = wsp + off;
    off += (bytes + 255) & ~(size_t)255;
    return p;
  };
  f16*   hall  = (f16*)alloc((size_t)B_ * LM_ * H_ * 2);
  f16*   wqh   = (f16*)alloc((size_t)H_ * H_ * 2);
  f16*   wkh   = (f16*)alloc((size_t)H_ * H_ * 2);
  f16*   wvh   = (f16*)alloc((size_t)H_ * H_ * 2);
  f16*   woh   = (f16*)alloc((size_t)H_ * H_ * 2);
  f16*   pench = (f16*)alloc((size_t)D_ * L_ * 2);
  f16*   qh    = (f16*)alloc((size_t)B_ * M_ * H_ * 2);
  f16*   kh    = (f16*)alloc((size_t)B_ * LM_ * H_ * 2);
  f16*   vth   = (f16*)alloc((size_t)B_ * NH_ * D_ * LM_ * 2);
  float* posq  = (float*)alloc((size_t)B_ * NH_ * M_ * L_ * 4);
  f16*   attnh = (f16*)alloc((size_t)B_ * M_ * H_ * 2);
  float* h1pre = (float*)alloc((size_t)B_ * M_ * H_ * 4);
  float* h1f   = (float*)alloc((size_t)B_ * M_ * H_ * 4);
  f16*   h1h   = (f16*)alloc((size_t)B_ * M_ * H_ * 2);
  float* gate  = (float*)alloc((size_t)NTOK_ * E_ * 4);
  float* moe   = (float*)alloc((size_t)NTOK_ * H_ * 4);
  f16*   tbuf  = (f16*)alloc((size_t)NTOK_ * F_ * 2);

  dim3 blk(256);
  // ---- f32 -> f16 conversions ----
  cvt_kernel<<<256, blk, 0, stream>>>(Wq, wqh, (size_t)H_ * H_);
  cvt_kernel<<<256, blk, 0, stream>>>(Wk, wkh, (size_t)H_ * H_);
  cvt_kernel<<<256, blk, 0, stream>>>(Wv, wvh, (size_t)H_ * H_);
  cvt_kernel<<<256, blk, 0, stream>>>(Wo, woh, (size_t)H_ * H_);
  cvt_kernel<<<64, blk, 0, stream>>>(pos_enc, pench, (size_t)D_ * L_);
  for (int b = 0; b < B_; ++b) {
    cvt_kernel<<<256, blk, 0, stream>>>(h_cache + (size_t)b * L_ * H_,
                                        hall + (size_t)b * LM_ * H_,
                                        (size_t)L_ * H_);
    cvt_kernel<<<256, blk, 0, stream>>>(h + (size_t)b * M_ * H_,
                                        hall + ((size_t)b * LM_ + L_) * H_,
                                        (size_t)M_ * H_);
  }
  // ---- K / V projections (all B*LM rows), V stored transposed ----
  gemm_kernel<0, f16><<<dim3(H_ / 128, (B_ * LM_) / 128), blk, 0, stream>>>(
      hall, H_, wkh, H_, nullptr, kh, H_, B_ * LM_, H_, H_, nullptr, nullptr,
      0, nullptr, 0);
  gemm_kernel<5, f16><<<dim3(H_ / 128, (B_ * LM_) / 128), blk, 0, stream>>>(
      hall, H_, wvh, H_, nullptr, vth, H_, B_ * LM_, H_, H_, nullptr, nullptr,
      0, nullptr, 0);
  // ---- Q projection (current tokens only, per batch) ----
  for (int b = 0; b < B_; ++b) {
    gemm_kernel<0, f16><<<dim3(H_ / 128, M_ / 128), blk, 0, stream>>>(
        hall + ((size_t)b * LM_ + L_) * H_, H_, wqh, H_, nullptr,
        qh + (size_t)b * M_ * H_, H_, M_, H_, H_, nullptr, nullptr, 0,
        nullptr, 0);
  }
  // ---- positional term: posq[bh] = q_head @ P  (f32 out) ----
  for (int bh = 0; bh < B_ * NH_; ++bh) {
    int b = bh / NH_, hd = bh % NH_;
    gemm_kernel<4, f16><<<dim3(L_ / 128, M_ / 128), blk, 0, stream>>>(
        qh + (size_t)b * M_ * H_ + hd * D_, H_, pench, L_,
        posq + (size_t)bh * M_ * L_, nullptr, L_, M_, L_, D_, nullptr,
        nullptr, 0, nullptr, 0);
  }
  // ---- sliding-window flash attention ----
  attn_kernel<<<(B_ * NH_ * (M_ / 16)) / 8, blk, 0, stream>>>(qh, kh, vth,
                                                              posq, attnh);
  // ---- output projection + residual ----
  gemm_kernel<1, f16><<<dim3(H_ / 128, (B_ * M_) / 128), blk, 0, stream>>>(
      attnh, H_, woh, H_, h1pre, nullptr, H_, B_ * M_, H_, H_, nullptr, h, H_,
      nullptr, 0);
  // ---- LN1 ----
  ln1_kernel<<<B_ * M_, blk, 0, stream>>>(h1pre, ln1_g, ln1_b, h1f, h1h);
  // ---- gating ----
  gate_kernel<<<NTOK_, blk, 0, stream>>>(h1f, gate_w, gate_b, gate);
  // ---- MoE ----
  zero_kernel<<<256, blk, 0, stream>>>(moe, (size_t)NTOK_ * H_);
  for (int e = 0; e < E_; ++e) {
    gemm_kernel<2, float><<<dim3(F_ / 128, NTOK_ / 128), blk, 0, stream>>>(
        h1h, H_, w1 + (size_t)e * H_ * F_, F_, nullptr, tbuf, F_, NTOK_, F_,
        H_, b1 + (size_t)e * F_, nullptr, 0, nullptr, 0);
    gemm_kernel<3, float><<<dim3(H_ / 128, NTOK_ / 128), blk, 0, stream>>>(
        tbuf, F_, w2 + (size_t)e * F_ * H_, H_, moe, nullptr, H_, NTOK_, H_,
        F_, b2 + (size_t)e * H_, nullptr, 0, gate + e, E_);
  }
  // ---- momentum + LN2 -> outputs ----
  float* out_h2  = (float*)d_out;
  float* out_mom = (float*)d_out + (size_t)B_ * M_ * H_;
  final_kernel<<<B_ * M_, blk, 0, stream>>>(h1f, momentum, moe, ln2_g, ln2_b,
                                            out_h2, out_mom);
}
